// CoPLGCF_36000415875265
// MI455X (gfx1250) — matrix operand order repaired and checked by reading, not compiled, not run on hardware
//
#include <hip/hip_runtime.h>

#define N_U 100000
#define N_I 50000
#define DIM 128
#define NLAYER 3
#define NNZ_POS 1000000
#define NNZ_NEG 500000
#define NNZ_II  500000
#define BATCH 8192
#define II_W 1.0f
#define LAMBDA_REG 1e-6f
#define NEG_SLOPE 0.2f

typedef __attribute__((ext_vector_type(16))) unsigned short v16u;
typedef __attribute__((ext_vector_type(8)))  unsigned short v8u;
typedef __attribute__((ext_vector_type(16))) __bf16         v16bf;
typedef __attribute__((ext_vector_type(8)))  float          v8f;

__device__ __forceinline__ unsigned short f2bf(float f) {
    unsigned u = __builtin_bit_cast(unsigned, f);
    u += 0x7FFFu + ((u >> 16) & 1u);   // round-to-nearest-even
    return (unsigned short)(u >> 16);
}

// ---------------------------------------------------------------------------
// COO SpMM scatter: Out[rows[e]] += vals[e] * X[cols[e]]   (one wave32 / edge)
// ---------------------------------------------------------------------------
__global__ __launch_bounds__(256) void spmm_scatter(
    const int* __restrict__ rows, const int* __restrict__ cols,
    const float* __restrict__ vals, const float* __restrict__ X,
    float* __restrict__ Out, int nnz) {
    int gid  = blockIdx.x * blockDim.x + threadIdx.x;
    int e    = gid >> 5;
    int lane = gid & 31;
    if (e >= nnz) return;
    int   r = rows[e], c = cols[e];
    float v = vals[e];
    float4 x = ((const float4*)(X + (size_t)c * DIM))[lane];
    float* dst = Out + (size_t)r * DIM + lane * 4;
    atomicAdd(dst + 0, v * x.x);
    atomicAdd(dst + 1, v * x.y);
    atomicAdd(dst + 2, v * x.z);
    atomicAdd(dst + 3, v * x.w);
}

// ---------------------------------------------------------------------------
// Pack fp32 weights W[k][e][d] into bf16 WMMA B-fragment layout.
// B (K x N) = W^T : B[K=k*128+d][N=e].
// Fragment index: (((ntile*Ksteps + s)*32 + lane)*16 + h)
//   K = s*32 + (lane>>4)*16 + h ; N = ntile*16 + (lane&15)
// ---------------------------------------------------------------------------
__global__ __launch_bounds__(256) void pack_w(
    const float* __restrict__ W, unsigned short* __restrict__ out, int nbranch) {
    int Ksteps = nbranch * 4;
    int total  = 8 * Ksteps * 32 * 16;
    int idx = blockIdx.x * blockDim.x + threadIdx.x;
    if (idx >= total) return;
    int h    = idx & 15;
    int lane = (idx >> 4) & 31;
    int rest = idx >> 9;
    int s    = rest % Ksteps;
    int t    = rest / Ksteps;
    int K  = s * 32 + (lane >> 4) * 16 + h;
    int n  = t * 16 + (lane & 15);
    int br = K >> 7, d = K & 127;
    out[idx] = f2bf(W[((size_t)br * 128 + n) * 128 + d]);
}

__global__ void bias_sum(const float* __restrict__ Wb, float* __restrict__ out,
                         int nbranch, float iiw) {
    int e = threadIdx.x;   // 128 threads
    float s = 0.f;
    for (int k = 0; k < nbranch; ++k)
        s += (k >= 5 ? iiw : 1.0f) * Wb[k * 128 + e];
    out[e] = s;
}

// ---------------------------------------------------------------------------
// Fused multi-branch dense layer:
//   Out = leakyrelu( sum_k branch_k @ W_k^T + bias )
// Block = 256 thr = 8 waves, M-tile = MT*16 rows -> MT accumulators per wave,
// MT WMMAs per B-fragment load (amortizes L2 weight traffic MT-fold).
// Wave w owns N-tile w (16 cols). Branch activations staged in LDS as bf16
// (row pitch 136 halves -> conflict-free ds_load_b128 across 16 rows).
// ---------------------------------------------------------------------------
template <int NBR, int MT>
__global__ __launch_bounds__(256) void dense_layer(
    const float* __restrict__ E,  const float* __restrict__ Zp,
    const float* __restrict__ Zn, const float* __restrict__ Zi,
    const unsigned short* __restrict__ packW, const float* __restrict__ biasSum,
    float* __restrict__ Out, int nrows, float iiw) {
    constexpr int ROWS = MT * 16;
    constexpr int BRS  = ROWS * 136;                    // halves per branch
    __shared__ __align__(16) unsigned short tile[NBR * BRS];
    const int row0 = blockIdx.x * ROWS;
    const int tid  = threadIdx.x;

    for (int idx = tid; idx < ROWS * 128; idx += 256) {
        int r = idx >> 7, c = idx & 127;
        int row = row0 + r;
        float e = 0.f, zp = 0.f, zn = 0.f, zi = 0.f;
        if (row < nrows) {
            size_t g = (size_t)row * DIM + c;
            e  = E[g];
            zp = Zp[g];
            zn = Zn[g];
            if (NBR == 7) zi = iiw * Zi[g];
        }
        int off = r * 136 + c;
        tile[0 * BRS + off] = f2bf(e);
        tile[1 * BRS + off] = f2bf(zp);
        tile[2 * BRS + off] = f2bf(zp * e);
        tile[3 * BRS + off] = f2bf(zn);
        tile[4 * BRS + off] = f2bf(zn * e);
        if (NBR == 7) {
            tile[5 * BRS + off] = f2bf(zi);
            tile[6 * BRS + off] = f2bf(zi * e);
        }
    }
    __syncthreads();

    const int lane = tid & 31;
    const int t    = tid >> 5;         // wave id == N-tile
    const int hsel = lane >> 4;        // lane group (K 0-15 vs 16-31 halves)
    const int m    = lane & 15;        // M row within 16-row sub-tile
    constexpr int Ksteps = NBR * 4;

    v8f acc[MT];
#pragma unroll
    for (int mt = 0; mt < MT; ++mt)
        acc[mt] = (v8f){0.f, 0.f, 0.f, 0.f, 0.f, 0.f, 0.f, 0.f};

    const unsigned short* bptr = packW + ((size_t)(t * Ksteps) * 32 + lane) * 16;

#pragma unroll
    for (int s = 0; s < Ksteps; ++s) {
        const int br    = s >> 2;
        const int kbase = (s & 3) * 32;
        // A fragment (matches packed B): lane<16 -> K {kb..kb+15}
        //                                lane>=16 -> K {kb+16..kb+31}
        const unsigned short* ap = &tile[br * BRS + m * 136 + kbase + hsel * 16];
        v16u B  = *(const v16u*)(bptr + (size_t)s * 32 * 16);
        v16bf Bb = __builtin_bit_cast(v16bf, B);
#pragma unroll
        for (int mt = 0; mt < MT; ++mt) {
            union { v16u v; v8u h[2]; } A;
            A.h[0] = *(const v8u*)(ap + mt * (16 * 136));
            A.h[1] = *(const v8u*)(ap + mt * (16 * 136) + 8);
            acc[mt] = __builtin_amdgcn_wmma_f32_16x16x32_bf16(
                false, __builtin_bit_cast(v16bf, A.v), false, Bb,
                (short)0, acc[mt], false, false);
        }
    }

    // C layout: VGPR r, lanes 0-15: (M=r, N=lane); lanes 16-31: (M=8+r, N=lane-16)
    int n = t * 16 + m;
    float bias = biasSum[n];
#pragma unroll
    for (int mt = 0; mt < MT; ++mt) {
#pragma unroll
        for (int r = 0; r < 8; ++r) {
            int row = row0 + mt * 16 + hsel * 8 + r;
            if (row < nrows) {
                float x = acc[mt][r] + bias;
                x = x > 0.f ? x : NEG_SLOPE * x;
                Out[(size_t)row * DIM + n] = x;
            }
        }
    }
}

// ---------------------------------------------------------------------------
// Final: normalize Eu rows for the batch, dot, BCE + reg loss.
// out[0] = loss, out[1..BATCH] = logits. One wave32 per batch element.
// ---------------------------------------------------------------------------
__global__ __launch_bounds__(256) void loss_kernel(
    const float* __restrict__ Eu, const float* __restrict__ Ei,
    const int* __restrict__ uids, const int* __restrict__ iids,
    const int* __restrict__ labels, float* __restrict__ out) {
    int gid  = blockIdx.x * blockDim.x + threadIdx.x;
    int b    = gid >> 5;
    int lane = gid & 31;
    if (b >= BATCH) return;
    float4 u = ((const float4*)(Eu + (size_t)uids[b] * DIM))[lane];
    float4 i = ((const float4*)(Ei + (size_t)iids[b] * DIM))[lane];
    float nu = u.x * u.x + u.y * u.y + u.z * u.z + u.w * u.w;
    float ni = i.x * i.x + i.y * i.y + i.z * i.z + i.w * i.w;
    float dt = u.x * i.x + u.y * i.y + u.z * i.z + u.w * i.w;
    for (int off = 16; off; off >>= 1) {
        nu += __shfl_xor(nu, off, 32);
        ni += __shfl_xor(ni, off, 32);
        dt += __shfl_xor(dt, off, 32);
    }
    if (lane == 0) {
        float inv   = 1.0f / fmaxf(sqrtf(nu), 1e-12f);
        float logit = dt * inv;
        out[1 + b] = logit;
        float y    = (float)labels[b];
        float bce  = fmaxf(logit, 0.f) - logit * y + log1pf(expf(-fabsf(logit)));
        float ureg = nu * inv * inv;   // ||u_emb||^2
        atomicAdd(out, bce * (1.0f / BATCH) + LAMBDA_REG * (ureg + ni));
    }
}

// ---------------------------------------------------------------------------
extern "C" void kernel_launch(void* const* d_in, const int* in_sizes, int n_in,
                              void* d_out, int out_size, void* d_ws, size_t ws_size,
                              hipStream_t stream) {
    (void)in_sizes; (void)n_in; (void)out_size; (void)ws_size;
    const float* Eu0  = (const float*)d_in[0];
    const float* Ei0  = (const float*)d_in[1];
    const float* Wu_w = (const float*)d_in[2];
    const float* Wu_b = (const float*)d_in[3];
    const float* Wi_w = (const float*)d_in[4];
    const float* Wi_b = (const float*)d_in[5];
    const int*   pr   = (const int*)d_in[6];
    const int*   pc   = (const int*)d_in[7];
    const float* pv   = (const float*)d_in[8];
    const int*   nr   = (const int*)d_in[9];
    const int*   nc   = (const int*)d_in[10];
    const float* nv   = (const float*)d_in[11];
    const int*   ir   = (const int*)d_in[12];
    const int*   ic   = (const int*)d_in[13];
    const float* iv   = (const float*)d_in[14];
    const int*   uids = (const int*)d_in[15];
    const int*   iids = (const int*)d_in[16];
    const int*   labels = (const int*)d_in[17];

    float* ws = (float*)d_ws;
    size_t o = 0;
    auto alloc = [&](size_t nf) { float* p = ws + o; o += nf; return p; };
    float* EuA = alloc((size_t)N_U * DIM);
    float* EuB = alloc((size_t)N_U * DIM);
    float* EiA = alloc((size_t)N_I * DIM);
    float* EiB = alloc((size_t)N_I * DIM);
    float* Zup = alloc((size_t)N_U * DIM);
    float* Zun = alloc((size_t)N_U * DIM);
    float* Zip = alloc((size_t)N_I * DIM);
    float* Zin = alloc((size_t)N_I * DIM);
    float* Zii = alloc((size_t)N_I * DIM);
    float* biasU = alloc(128);
    float* biasI = alloc(128);
    unsigned short* packU = (unsigned short*)alloc(8 * 20 * 32 * 16 / 2);  // 81920 bf16
    unsigned short* packI = (unsigned short*)alloc(8 * 28 * 32 * 16 / 2);  // 114688 bf16

    const float* Eu = Eu0;
    const float* Ei = Ei0;
    float* EuN = EuA;
    float* EiN = EiA;

    constexpr int ROWS  = 64;                       // MT = 4
    const int gridU = (N_U + ROWS - 1) / ROWS;      // 1563 (last tile partial)
    const int gridI = (N_I + ROWS - 1) / ROWS;      // 782  (last tile partial)

    for (int l = 0; l < NLAYER; ++l) {
        hipMemsetAsync(Zup, 0, (size_t)N_U * DIM * 4, stream);
        hipMemsetAsync(Zun, 0, (size_t)N_U * DIM * 4, stream);
        hipMemsetAsync(Zip, 0, (size_t)N_I * DIM * 4, stream);
        hipMemsetAsync(Zin, 0, (size_t)N_I * DIM * 4, stream);
        hipMemsetAsync(Zii, 0, (size_t)N_I * DIM * 4, stream);

        spmm_scatter<<<NNZ_POS / 8, 256, 0, stream>>>(pr, pc, pv, Ei, Zup, NNZ_POS);
        spmm_scatter<<<NNZ_NEG / 8, 256, 0, stream>>>(nr, nc, nv, Ei, Zun, NNZ_NEG);
        spmm_scatter<<<NNZ_POS / 8, 256, 0, stream>>>(pc, pr, pv, Eu, Zip, NNZ_POS);
        spmm_scatter<<<NNZ_NEG / 8, 256, 0, stream>>>(nc, nr, nv, Eu, Zin, NNZ_NEG);
        spmm_scatter<<<NNZ_II  / 8, 256, 0, stream>>>(ir, ic, iv, Ei, Zii, NNZ_II);

        pack_w<<<(5 * 16384 + 255) / 256, 256, 0, stream>>>(
            Wu_w + (size_t)l * 5 * 128 * 128, packU, 5);
        pack_w<<<(7 * 16384 + 255) / 256, 256, 0, stream>>>(
            Wi_w + (size_t)l * 7 * 128 * 128, packI, 7);
        bias_sum<<<1, 128, 0, stream>>>(Wu_b + (size_t)l * 5 * 128, biasU, 5, 1.0f);
        bias_sum<<<1, 128, 0, stream>>>(Wi_b + (size_t)l * 7 * 128, biasI, 7, II_W);

        dense_layer<5, 4><<<gridU, 256, 0, stream>>>(Eu, Zup, Zun, nullptr,
                                                     packU, biasU, EuN, N_U, 1.0f);
        dense_layer<7, 4><<<gridI, 256, 0, stream>>>(Ei, Zip, Zin, Zii,
                                                     packI, biasI, EiN, N_I, II_W);
        Eu = EuN; Ei = EiN;
        EuN = (EuN == EuA) ? EuB : EuA;
        EiN = (EiN == EiA) ? EiB : EiA;
    }

    hipMemsetAsync(d_out, 0, sizeof(float), stream);
    loss_kernel<<<BATCH / 8, 256, 0, stream>>>(Eu, Ei, uids, iids, labels,
                                               (float*)d_out);
}
